// cross_modal_hypergraph_62612033241631
// MI455X (gfx1250) — matrix operand order repaired and checked by reading, not compiled, not run on hardware
//
#include <hip/hip_runtime.h>
#include <hip/hip_bf16.h>

// Problem constants (match reference)
#define DD   64     // dialogues
#define LL   64     // utterances per dialogue
#define MM   192    // nodes (=hyperedges) per dialogue
#define DIMF 512    // feature dim
#define GG   256    // graph dim
#define NN   4096   // D*L
#define KTOP 12     // int(L * 0.2)
#define TAUI 14.285714285714286f  // 1/0.07

typedef __attribute__((ext_vector_type(16))) __bf16 v16bf;
typedef __attribute__((ext_vector_type(8)))  float  v8f;

// ---------------- WMMA helpers (CDNA5 wave32, bf16 -> f32 accum) ----------------

__device__ __forceinline__ v8f wmma_bf16(v16bf a, v16bf b, v8f c) {
  // D = A(16x32) x B(32x16) + C(16x16 f32)
  return __builtin_amdgcn_wmma_f32_16x16x32_bf16(
      /*neg_a=*/false, a, /*neg_b=*/false, b,
      /*c_mod=*/(short)0, c, /*reuse_a=*/false, /*reuse_b=*/false);
}

// A tile 16x32 (MxK), row-major bf16 source with leading dim ld.
// Lanes 0-15: M=lane, K=0..15; lanes 16-31: M=lane-16, K=16..31.
__device__ __forceinline__ v16bf load_a_bf16(const __bf16* base, int ld) {
  int lane = threadIdx.x & 31;
  int row  = lane & 15;
  int hi   = lane >> 4;
  const __bf16* p = base + (size_t)row * ld + hi * 16;
  v16bf a;
#pragma unroll
  for (int i = 0; i < 16; ++i) a[i] = p[i];
  return a;
}

// B tile 32x16 (KxN) loaded from a row-major "B^T" operand Bt[n][k] with leading dim ld
// (i.e. computes A @ Bt^T). Lanes 0-15: N=lane, K=0..15; lanes 16-31: N=lane-16, K=16..31.
__device__ __forceinline__ v16bf load_bt_bf16(const __bf16* base, int ld) {
  int lane = threadIdx.x & 31;
  int col  = lane & 15;
  int kb   = (lane >> 4) * 16;
  const __bf16* p = base + (size_t)col * ld + kb;
  v16bf b;
#pragma unroll
  for (int i = 0; i < 16; ++i) b[i] = p[i];
  return b;
}

// C/D layout: element r of v8f -> row M = r + 8*(lane>>4), col N = lane&15.

// ---------------- Stage kernels ----------------

// Zero loss accumulators (rowsum[6*4096] and diag[1]).
__global__ void zero_kernel(float* rowsum, float* diag) {
  int t = blockIdx.x * blockDim.x + threadIdx.x;
  if (t < 6 * NN) rowsum[t] = 0.0f;
  if (t == 0) diag[0] = 0.0f;
}

// Transpose+convert weights to bf16: Wfc_t[g][k], Wh_t[g][k].
__global__ void wtrans_kernel(const float* W_fc, const float* W_h,
                              __bf16* Wfc_tb, __bf16* Wh_tb) {
  int t = blockIdx.x * blockDim.x + threadIdx.x;
  if (t < DIMF * GG) {
    int k = t / GG, g = t % GG;
    Wfc_tb[(size_t)g * DIMF + k] = (__bf16)W_fc[t];
  }
  if (t < GG * GG) {
    int k = t / GG, g = t % GG;
    Wh_tb[(size_t)g * GG + k] = (__bf16)W_h[t];
  }
}

// Build featb[d][m][dim] (bf16) from t/a/v and per-node squared norms (f32).
__global__ void prep_kernel(const float* tt, const float* aa, const float* vv,
                            __bf16* featb, float* sq) {
  int row = blockIdx.x;                  // 0..12287 = d*192+m
  int d = row / MM, m = row % MM;
  const float* src;
  if (m < LL)           src = tt + ((size_t)d * LL + m) * DIMF;
  else if (m < 2 * LL)  src = aa + ((size_t)d * LL + (m - LL)) * DIMF;
  else                  src = vv + ((size_t)d * LL + (m - 2 * LL)) * DIMF;
  __bf16* dst = featb + (size_t)row * DIMF;
  __shared__ float red[256];
  float ss = 0.0f;
  for (int c = threadIdx.x; c < DIMF; c += 256) {
    float x = src[c];
    dst[c] = (__bf16)x;
    ss += x * x;
  }
  red[threadIdx.x] = ss; __syncthreads();
  for (int s = 128; s > 0; s >>= 1) {
    if ((int)threadIdx.x < s) red[threadIdx.x] += red[threadIdx.x + s];
    __syncthreads();
  }
  if (threadIdx.x == 0) sq[row] = red[0];
}

// Per-dialogue squared-distance matrix via WMMA Gram: d2 = max(sq_i+sq_j-2*feat@feat^T, 0).
// One wave per 16x16 tile; 64 dialogues * 12 * 12 tiles = 9216 waves.
__global__ void gram_d2_kernel(const __bf16* featb, const float* sq, float* d2) {
  int wid  = (blockIdx.x * blockDim.x + threadIdx.x) >> 5;
  int lane = threadIdx.x & 31;
  int d  = wid / 144;
  int r2 = wid % 144;
  int ti = r2 / 12, tj = r2 % 12;
  const __bf16* A = featb + ((size_t)d * MM + ti * 16) * DIMF;
  const __bf16* B = featb + ((size_t)d * MM + tj * 16) * DIMF;
  v8f acc = {};
  for (int k = 0; k < DIMF; k += 32)
    acc = wmma_bf16(load_a_bf16(A + k, DIMF), load_bt_bf16(B + k, DIMF), acc);
  int n = lane & 15, mb = (lane >> 4) * 8;
  const float* sqd = sq + (size_t)d * MM;
#pragma unroll
  for (int r = 0; r < 8; ++r) {
    int i = ti * 16 + mb + r;
    int j = tj * 16 + n;
    float v = sqd[i] + sqd[j] - 2.0f * acc[r];
    d2[((size_t)d * MM + i) * MM + j] = fmaxf(v, 0.0f);
  }
}

// Top-12 nearest + central nodes -> H (bf16 0/1), H^T, deg_e. One thread per hyperedge.
__global__ void topk_h_kernel(const float* d2, __bf16* Hb, __bf16* Htb, float* deg_e) {
  int t = blockIdx.x * blockDim.x + threadIdx.x;
  if (t >= DD * MM) return;
  int d = t / MM, e = t % MM;
  const float* row = d2 + ((size_t)d * MM + e) * MM;
  unsigned mask[6] = {0, 0, 0, 0, 0, 0};
  float prevv = -1.0f; int previ = -1;
  for (int s = 0; s < KTOP; ++s) {   // exact lexicographic top-k smallest
    float best = 3.4e38f; int bi = 0;
    for (int j = 0; j < MM; ++j) {
      float v = row[j];
      bool after = (v > prevv) || (v == prevv && j > previ);
      if (after && v < best) { best = v; bi = j; }
    }
    prevv = best; previ = bi;
    mask[bi >> 5] |= 1u << (bi & 31);
  }
  int u = e / 3;                      // central utterance of edge e
  mask[u >> 5]           |= 1u << (u & 31);
  mask[(u + LL) >> 5]    |= 1u << ((u + LL) & 31);
  mask[(u + 2*LL) >> 5]  |= 1u << ((u + 2*LL) & 31);
  int cnt = 0;
  __bf16* hrow = Hb  + ((size_t)d * MM + e) * MM;
  __bf16* hcol = Htb + (size_t)d * MM * MM + e;
  for (int j = 0; j < MM; ++j) {
    int on = (mask[j >> 5] >> (j & 31)) & 1;
    cnt += on;
    __bf16 val = (__bf16)(float)on;
    hrow[j] = val;
    hcol[(size_t)j * MM] = val;
  }
  deg_e[t] = (float)cnt;
}

// deg_v[n] = sum over edges of H[e][n] (row sums of H^T).
__global__ void degv_kernel(const __bf16* Htb, float* deg_v) {
  int t = blockIdx.x * blockDim.x + threadIdx.x;
  if (t >= DD * MM) return;
  const __bf16* r = Htb + (size_t)t * MM;
  float s = 0.0f;
  for (int j = 0; j < MM; ++j) s += (float)r[j];
  deg_v[t] = s;
}

// X = feat @ W_fc + b_fc, stored transposed-bf16: Xt[d][g][m].
__global__ void xgemm_kernel(const __bf16* featb, const __bf16* Wfc_tb,
                             const float* b_fc, __bf16* Xt_b) {
  int wid  = (blockIdx.x * blockDim.x + threadIdx.x) >> 5;
  int lane = threadIdx.x & 31;
  int rt = wid >> 4, gt = wid & 15;          // 768 row tiles x 16 col tiles
  const __bf16* A = featb  + (size_t)rt * 16 * DIMF;
  const __bf16* B = Wfc_tb + (size_t)gt * 16 * DIMF;
  v8f acc = {};
  for (int k = 0; k < DIMF; k += 32)
    acc = wmma_bf16(load_a_bf16(A + k, DIMF), load_bt_bf16(B + k, DIMF), acc);
  int n = lane & 15, mb = (lane >> 4) * 8;
  int g = gt * 16 + n;
  float bias = b_fc[g];
#pragma unroll
  for (int r = 0; r < 8; ++r) {
    int R = rt * 16 + mb + r;
    int d = R / MM, mloc = R % MM;
    Xt_b[((size_t)d * GG + g) * MM + mloc] = (__bf16)(acc[r] + bias);
  }
}

// E = (H @ X) / deg_e, stored transposed-bf16: Et[d][g][e].
__global__ void egemm_kernel(const __bf16* Hb, const __bf16* Xt_b,
                             const float* deg_e, __bf16* Et_b) {
  int wid  = (blockIdx.x * blockDim.x + threadIdx.x) >> 5;
  int lane = threadIdx.x & 31;
  int gt = wid & 15; int tmp = wid >> 4;
  int et = tmp % 12; int d = tmp / 12;       // 64 * 12 * 16 = 12288 waves
  const __bf16* A = Hb   + ((size_t)d * MM + et * 16) * MM;
  const __bf16* B = Xt_b + ((size_t)d * GG + gt * 16) * MM;
  v8f acc = {};
  for (int k = 0; k < MM; k += 32)
    acc = wmma_bf16(load_a_bf16(A + k, MM), load_bt_bf16(B + k, MM), acc);
  int n = lane & 15, mb = (lane >> 4) * 8;
  int g = gt * 16 + n;
#pragma unroll
  for (int r = 0; r < 8; ++r) {
    int e = et * 16 + mb + r;
    float v = acc[r] / deg_e[d * MM + e];
    Et_b[((size_t)d * GG + g) * MM + e] = (__bf16)v;
  }
}

// Y = (H^T @ E) / deg_v, stored row-major bf16: Yb[d][n][g].
__global__ void ygemm_kernel(const __bf16* Htb, const __bf16* Et_b,
                             const float* deg_v, __bf16* Yb) {
  int wid  = (blockIdx.x * blockDim.x + threadIdx.x) >> 5;
  int lane = threadIdx.x & 31;
  int gt = wid & 15; int tmp = wid >> 4;
  int nt = tmp % 12; int d = tmp / 12;
  const __bf16* A = Htb  + ((size_t)d * MM + nt * 16) * MM;
  const __bf16* B = Et_b + ((size_t)d * GG + gt * 16) * MM;
  v8f acc = {};
  for (int k = 0; k < MM; k += 32)
    acc = wmma_bf16(load_a_bf16(A + k, MM), load_bt_bf16(B + k, MM), acc);
  int n = lane & 15, mb = (lane >> 4) * 8;
  int g = gt * 16 + n;
#pragma unroll
  for (int r = 0; r < 8; ++r) {
    int nd = nt * 16 + mb + r;
    float v = acc[r] / deg_v[d * MM + nd];
    Yb[((size_t)d * MM + nd) * GG + g] = (__bf16)v;
  }
}

// out = relu(Y @ W_h + b_h), scattered into out_cat[N][768] (reverse_features layout).
__global__ void outgemm_kernel(const __bf16* Yb, const __bf16* Wh_tb,
                               const float* b_h, float* outcat) {
  int wid  = (blockIdx.x * blockDim.x + threadIdx.x) >> 5;
  int lane = threadIdx.x & 31;
  int rt = wid >> 4, gt = wid & 15;          // 768 x 16 tiles
  const __bf16* A = Yb    + (size_t)rt * 16 * GG;
  const __bf16* B = Wh_tb + (size_t)gt * 16 * GG;
  v8f acc = {};
  for (int k = 0; k < GG; k += 32)
    acc = wmma_bf16(load_a_bf16(A + k, GG), load_bt_bf16(B + k, GG), acc);
  int n = lane & 15, mb = (lane >> 4) * 8;
  int g = gt * 16 + n;
  float bias = b_h[g];
#pragma unroll
  for (int r = 0; r < 8; ++r) {
    int R = rt * 16 + mb + r;
    int d = R / MM, nloc = R % MM;
    int blk = nloc / LL, u = nloc % LL;
    float v = fmaxf(acc[r] + bias, 0.0f);
    outcat[((size_t)(d * LL + u)) * (3 * GG) + blk * GG + g] = v;
  }
}

// Row-normalize each modality segment of out_cat into nrm[3][N][G] (bf16).
__global__ void norm_kernel(const float* outcat, __bf16* nrm) {
  int b = blockIdx.x;                 // 0..12287 = p*4096 + row
  int p = b / NN, r = b % NN;
  int c = threadIdx.x;                // 256
  float v = outcat[(size_t)r * (3 * GG) + p * GG + c];
  __shared__ float red[256];
  red[c] = v * v; __syncthreads();
  for (int s = 128; s > 0; s >>= 1) {
    if (c < s) red[c] += red[c + s];
    __syncthreads();
  }
  float inv = v / (sqrtf(red[0]) + 1e-8f);
  nrm[((size_t)p * NN + r) * GG + c] = (__bf16)inv;
}

// Flash-style contrastive pass: for ordered pair (px,py), rows of sim = (xn_px @ xn_py^T)/tau.
// A fragments for the whole 16-row strip (K=256) are preloaded into registers once
// (8 x v16bf = 64 VGPRs/lane) and reused across all 16 column tiles: 16x less A traffic,
// 8 back-to-back WMMAs per tile against contiguous 128-bit B loads.
// Accumulates rowsum[pair][i] += sum_j exp(sim_ij - 1/tau) and diag[0] += sim_ii.
// Grid: (256 row tiles, 16 col chunks of 256 cols, 6 pairs), one wave per block.
__global__ void loss_tiles_kernel(const __bf16* nrm, float* rowsum, float* diag) {
  const int PX[6] = {0, 1, 0, 2, 1, 2};
  const int PY[6] = {1, 0, 2, 0, 2, 1};
  int lane  = threadIdx.x;
  int rt    = blockIdx.x;
  int chunk = blockIdx.y;
  int pair  = blockIdx.z;
  const __bf16* X = nrm + ((size_t)PX[pair] * NN + rt * 16) * GG;
  const __bf16* Ybase = nrm + (size_t)PY[pair] * NN * GG;
  int n = lane & 15, mb = (lane >> 4) * 8;

  // Preload full A strip: 16 rows x K=256 -> 8 fragments held in registers.
  v16bf af[8];
#pragma unroll
  for (int kk = 0; kk < 8; ++kk) af[kk] = load_a_bf16(X + kk * 32, GG);

  float rs[8] = {0, 0, 0, 0, 0, 0, 0, 0};
  float dsum = 0.0f;
  for (int ct = 0; ct < 16; ++ct) {
    int c0 = (chunk * 16 + ct) * 16;
    const __bf16* Y = Ybase + (size_t)c0 * GG;
    v8f acc = {};
#pragma unroll
    for (int kk = 0; kk < 8; ++kk)
      acc = wmma_bf16(af[kk], load_bt_bf16(Y + kk * 32, GG), acc);
    bool isdiag = (c0 == rt * 16);
#pragma unroll
    for (int r = 0; r < 8; ++r) {
      float s = acc[r] * TAUI;                 // sim = dot / tau
      rs[r] += __expf(s - TAUI);               // shift by max bound 1/tau
      if (isdiag && (mb + r) == n) dsum += s;
    }
  }
  // per-row reduction across the 16 lanes holding each row
#pragma unroll
  for (int r = 0; r < 8; ++r) {
    float v = rs[r];
    for (int off = 8; off >= 1; off >>= 1) v += __shfl_xor(v, off, 16);
    if (n == 0)
      atomicAdd(&rowsum[(size_t)pair * NN + rt * 16 + mb + r], v);
  }
  for (int off = 16; off >= 1; off >>= 1) dsum += __shfl_xor(dsum, off, 32);
  if (lane == 0 && dsum != 0.0f) atomicAdd(diag, dsum);
}

// loss = (sum over 6 ordered pairs,i of lse - sum of diag) / (6*N)
__global__ void loss_final_kernel(const float* rowsum, const float* diag, float* loss_out) {
  __shared__ float red[256];
  int tid = threadIdx.x;
  float s = 0.0f;
  for (int i = tid; i < 6 * NN; i += 256) s += logf(rowsum[i]) + TAUI;
  red[tid] = s; __syncthreads();
  for (int st = 128; st > 0; st >>= 1) {
    if (tid < st) red[tid] += red[tid + st];
    __syncthreads();
  }
  if (tid == 0) loss_out[0] = (red[0] - diag[0]) / (6.0f * (float)NN);
}

// ---------------- Host launch ----------------

extern "C" void kernel_launch(void* const* d_in, const int* in_sizes, int n_in,
                              void* d_out, int out_size, void* d_ws, size_t ws_size,
                              hipStream_t stream) {
  (void)in_sizes; (void)n_in; (void)out_size; (void)ws_size;
  const float* t_in  = (const float*)d_in[0];
  const float* a_in  = (const float*)d_in[1];
  const float* v_in  = (const float*)d_in[2];
  const float* W_fc  = (const float*)d_in[3];
  const float* b_fc  = (const float*)d_in[4];
  const float* W_h   = (const float*)d_in[5];
  const float* b_h   = (const float*)d_in[6];
  float* outcat = (float*)d_out;                     // [4096][768]
  float* loss   = outcat + (size_t)NN * 3 * GG;      // final scalar

  // Workspace carve-up (256B aligned)
  char* p = (char*)d_ws;
  auto take = [&](size_t bytes) { char* r = p; p += (bytes + 255) & ~(size_t)255; return r; };
  __bf16* featb  = (__bf16*)take((size_t)DD * MM * DIMF * 2);
  float*  sq     = (float*) take((size_t)DD * MM * 4);
  __bf16* Wfc_tb = (__bf16*)take((size_t)GG * DIMF * 2);
  __bf16* Wh_tb  = (__bf16*)take((size_t)GG * GG * 2);
  float*  d2     = (float*) take((size_t)DD * MM * MM * 4);
  __bf16* Hb     = (__bf16*)take((size_t)DD * MM * MM * 2);
  __bf16* Htb    = (__bf16*)take((size_t)DD * MM * MM * 2);
  float*  deg_e  = (float*) take((size_t)DD * MM * 4);
  float*  deg_v  = (float*) take((size_t)DD * MM * 4);
  __bf16* Xt_b   = (__bf16*)take((size_t)DD * GG * MM * 2);
  __bf16* Et_b   = (__bf16*)take((size_t)DD * GG * MM * 2);
  __bf16* Yb     = (__bf16*)take((size_t)DD * MM * GG * 2);
  __bf16* nrm    = (__bf16*)take((size_t)3 * NN * GG * 2);
  float*  rowsum = (float*) take((size_t)6 * NN * 4);
  float*  diag   = (float*) take(4);

  zero_kernel<<<96, 256, 0, stream>>>(rowsum, diag);
  wtrans_kernel<<<512, 256, 0, stream>>>(W_fc, W_h, Wfc_tb, Wh_tb);
  prep_kernel<<<DD * MM, 256, 0, stream>>>(t_in, a_in, v_in, featb, sq);
  gram_d2_kernel<<<1152, 256, 0, stream>>>(featb, sq, d2);          // 9216 waves
  topk_h_kernel<<<48, 256, 0, stream>>>(d2, Hb, Htb, deg_e);
  degv_kernel<<<48, 256, 0, stream>>>(Htb, deg_v);
  xgemm_kernel<<<1536, 256, 0, stream>>>(featb, Wfc_tb, b_fc, Xt_b); // 12288 waves
  egemm_kernel<<<1536, 256, 0, stream>>>(Hb, Xt_b, deg_e, Et_b);
  ygemm_kernel<<<1536, 256, 0, stream>>>(Htb, Et_b, deg_v, Yb);
  outgemm_kernel<<<1536, 256, 0, stream>>>(Yb, Wh_tb, b_h, outcat);
  norm_kernel<<<3 * NN, 256, 0, stream>>>(outcat, nrm);
  dim3 lgrid(NN / 16, 16, 6);
  loss_tiles_kernel<<<lgrid, 32, 0, stream>>>(nrm, rowsum, diag);
  loss_final_kernel<<<1, 256, 0, stream>>>(rowsum, diag, loss);
}
